// HyperEncoder_24592982737021
// MI455X (gfx1250) — matrix operand order repaired and checked by reading, not compiled, not run on hardware
//
#include <hip/hip_runtime.h>
#include <math.h>

typedef __attribute__((ext_vector_type(8)))  __bf16 v8bf;
typedef __attribute__((ext_vector_type(16))) __bf16 v16bf;
typedef __attribute__((ext_vector_type(8)))  float  v8f;

union BF16x16 { v16bf v; v8bf h[2]; };

#define D 128          // COMM_DIM
#define BATCH 128
#define MSG 16

// Fused hypernetwork GEMM.
// grid = 128 workgroups: g = blk/4 -> 4 batches [4g,4g+4); q = blk%4 -> k-quarter.
// block = 256 threads = 8 waves; wave w owns dd-block [16w,16w+16) for its 64 k values.
// Z^T tiles via v_wmma_f32_16x16x32_bf16 with bf16 hi/lo split (~f32 accuracy, 3 products
// accumulated into one f32 accumulator), folded with x_k and Wb into y, combined across
// k-quarters with global_atomic_add_f32.
__global__ __launch_bounds__(256) void hyper_main(
    const float* __restrict__ messages,  // [128][16][128]
    const float* __restrict__ Ww,        // [256][32768]
    const float* __restrict__ Wb,        // [32768]
    float* __restrict__ tmpY)            // [128][16][128] (pre-zeroed)
{
    __shared__ __align__(32) __bf16 s_xhi[64 * 256];    // 4 b-tiles * 16 rows * 256 K (32 KB)
    __shared__ __align__(32) __bf16 s_xlo[64 * 256];    // 32 KB
    __shared__ __align__(16) __bf16 s_whi[8][16 * 64];  // per-wave W^T quarter chunk (16 KB tot)
    __shared__ __align__(16) __bf16 s_wlo[8][16 * 64];  // 16 KB tot
    __shared__ float s_agg[4 * 128];                    // 2 KB

    const int tid  = threadIdx.x;
    const int w    = tid >> 5;          // wave id 0..7
    const int lane = tid & 31;
    const int g    = blockIdx.x >> 2;   // batch group 0..31
    const int q    = blockIdx.x & 3;    // k-quarter 0..3

    // ---- Stage X = [messages | agg] as bf16 hi/lo into LDS ----
    for (int idx = tid; idx < 4 * 128; idx += 256) {
        const int bt = idx >> 7, dd = idx & 127;
        const float* mp = messages + (size_t)(4 * g + bt) * MSG * D + dd;
        float s = 0.f;
        #pragma unroll
        for (int m = 0; m < MSG; ++m) s += mp[m * D];
        s_agg[idx] = s;
    }
    __syncthreads();
    for (int idx = tid; idx < 64 * 256; idx += 256) {
        const int row = idx >> 8, j = idx & 255;
        const int bt = row >> 4, m = row & 15;
        float val = (j < D)
            ? messages[(size_t)((4 * g + bt) * MSG + m) * D + j]
            : s_agg[bt * 128 + (j - D)];
        __bf16 h = (__bf16)val;
        s_xhi[idx] = h;
        s_xlo[idx] = (__bf16)(val - (float)h);
    }
    __syncthreads();

    const int r  = lane & 15;           // output row within b-tile (N of Z^T)
    const int hb = lane >> 4;           // lane half
    const int ci = lane & 15;           // W column owned during staging / A-operand M
    const int abase = ci * 64 + hb * 8; // A-operand base (halfwords) within quarter chunk

    v8f yacc[4] = {};

    for (int k = 0; k < 64; ++k) {
        const int kk = q * 64 + k;              // global k index 0..255
        const int c0 = kk * D + w * 16;         // global Ww column base for this wave

        v8f zacc[4] = {};

        #pragma unroll 1
        for (int jq = 0; jq < 4; ++jq) {        // j in [64*jq, 64*jq+64)
            // ---- stage W^T quarter chunk (wave-private; same-wave DS ordering) ----
            #pragma unroll 4
            for (int t = 0; t < 16; ++t) {
                const int p  = hb + 2 * t;      // pair index 0..31
                const int jl = 2 * p;           // local j (even), 0..62
                const size_t jg = (size_t)(jq * 64 + jl);
                const float* src = Ww + jg * 32768 + c0 + ci;
                const float f0 = src[0];
                const float f1 = src[32768];
                const __bf16 h0 = (__bf16)f0, h1 = (__bf16)f1;
                s_whi[w][ci * 64 + jl]     = h0;
                s_whi[w][ci * 64 + jl + 1] = h1;
                s_wlo[w][ci * 64 + jl]     = (__bf16)(f0 - (float)h0);
                s_wlo[w][ci * 64 + jl + 1] = (__bf16)(f1 - (float)h1);
            }
            // ---- WMMA over this quarter (2 k-steps of 32) ----
            #pragma unroll 1
            for (int s = 0; s < 2; ++s) {
                const int aoff = abase + s * 32;
                BF16x16 Ahi, Alo;
                Ahi.h[0] = *(const v8bf*)&s_whi[w][aoff];
                Ahi.h[1] = *(const v8bf*)&s_whi[w][aoff + 16];
                Alo.h[0] = *(const v8bf*)&s_wlo[w][aoff];
                Alo.h[1] = *(const v8bf*)&s_wlo[w][aoff + 16];
                const int xoff0 = hb * 8 + jq * 64 + s * 32;
                #pragma unroll
                for (int bt = 0; bt < 4; ++bt) {
                    const int xoff = (bt * 16 + r) * 256 + xoff0;
                    BF16x16 Bhi, Blo;
                    Bhi.h[0] = *(const v8bf*)&s_xhi[xoff];
                    Bhi.h[1] = *(const v8bf*)&s_xhi[xoff + 16];
                    Blo.h[0] = *(const v8bf*)&s_xlo[xoff];
                    Blo.h[1] = *(const v8bf*)&s_xlo[xoff + 16];
                    zacc[bt] = __builtin_amdgcn_wmma_f32_16x16x32_bf16(
                        false, Ahi.v, false, Bhi.v, (short)0, zacc[bt], false, false);
                    zacc[bt] = __builtin_amdgcn_wmma_f32_16x16x32_bf16(
                        false, Ahi.v, false, Blo.v, (short)0, zacc[bt], false, false);
                    zacc[bt] = __builtin_amdgcn_wmma_f32_16x16x32_bf16(
                        false, Alo.v, false, Bhi.v, (short)0, zacc[bt], false, false);
                }
            }
        }

        // ---- fold Z (+Wb) with x_k into y accumulators ----
        float wbv[8];
        {
            const float* wp = Wb + c0 + hb * 8;
            #pragma unroll
            for (int v = 0; v < 8; ++v) wbv[v] = wp[v];
        }
        #pragma unroll
        for (int bt = 0; bt < 4; ++bt) {
            const int xi = (bt * 16 + r) * 256 + kk;
            const float xk = (float)s_xhi[xi] + (float)s_xlo[xi];
            #pragma unroll
            for (int v = 0; v < 8; ++v)
                yacc[bt][v] += xk * (zacc[bt][v] + wbv[v]);
        }
    }

    // ---- combine k-quarters ----
    #pragma unroll
    for (int bt = 0; bt < 4; ++bt) {
        float* dst = tmpY + (size_t)((4 * g + bt) * MSG + r) * D + w * 16 + hb * 8;
        #pragma unroll
        for (int v = 0; v < 8; ++v)
            __hip_atomic_fetch_add(&dst[v], yacc[bt][v],
                                   __ATOMIC_RELAXED, __HIP_MEMORY_SCOPE_AGENT);
    }
}

// Epilogue: bias = x @ bw + bb, ELU, reduce over m. grid=128 (one b), block=128 (one dd).
__global__ __launch_bounds__(128) void hyper_epilogue(
    const float* __restrict__ messages,
    const float* __restrict__ bw,    // [256]
    const float* __restrict__ bb,    // [1]
    const float* __restrict__ tmpY,  // [128][16][128]
    float* __restrict__ out)         // [128][1][128]
{
    __shared__ float aggL[128];
    __shared__ float biasL[16];
    const int b = blockIdx.x, t = threadIdx.x;
    const float* mb = messages + (size_t)b * MSG * D;

    float s = 0.f;
    #pragma unroll
    for (int m = 0; m < MSG; ++m) s += mb[m * D + t];
    aggL[t] = s;
    __syncthreads();

    if (t < MSG) {
        float bias = bb[0];
        for (int j = 0; j < D; ++j) bias += mb[t * D + j] * bw[j];
        for (int j = 0; j < D; ++j) bias += aggL[j] * bw[D + j];
        biasL[t] = bias;
    }
    __syncthreads();

    float acc = 0.f;
    #pragma unroll
    for (int m = 0; m < MSG; ++m) {
        float v = tmpY[(size_t)(b * MSG + m) * D + t] + biasL[m];
        acc += (v > 0.f) ? v : expm1f(v);   // ELU(alpha=1)
    }
    out[(size_t)b * D + t] = acc;
}

extern "C" void kernel_launch(void* const* d_in, const int* in_sizes, int n_in,
                              void* d_out, int out_size, void* d_ws, size_t ws_size,
                              hipStream_t stream) {
    (void)in_sizes; (void)n_in; (void)out_size; (void)ws_size;
    const float* messages = (const float*)d_in[0];
    const float* Ww       = (const float*)d_in[1];
    const float* Wb       = (const float*)d_in[2];
    const float* bw       = (const float*)d_in[3];
    const float* bb       = (const float*)d_in[4];
    float* tmpY = (float*)d_ws;                       // 128*16*128 floats = 1 MB

    hipMemsetAsync(tmpY, 0, (size_t)BATCH * MSG * D * sizeof(float), stream);
    hyper_main<<<128, 256, 0, stream>>>(messages, Ww, Wb, tmpY);
    hyper_epilogue<<<128, 128, 0, stream>>>(messages, bw, bb, tmpY, (float*)d_out);
}
